// VectorQuantizer_19155554140247
// MI455X (gfx1250) — matrix operand order repaired and checked by reading, not compiled, not run on hardware
//
#include <hip/hip_runtime.h>
#include <hip/hip_bf16.h>
#include <stdint.h>

// ---------------------------------------------------------------------------
// VQ-VAE vector quantizer for MI455X (gfx1250, wave32, WMMA).
// Distance argmin via bf16 hi/lo split GEMM on v_wmma_f32_16x16x32_bf16,
// code tiles streamed to LDS by the Tensor Data Mover (double buffered).
// ---------------------------------------------------------------------------

#define NTOK   16384
#define NCODE  8192
#define DIM    256
#define BATCH  16
#define HW     1024
#define NELEM  (BATCH * DIM * HW)     // 4194304
#define USE_TDM 1

typedef __attribute__((ext_vector_type(16))) __bf16    v16bf;
typedef __attribute__((ext_vector_type(8)))  float     v8f;
typedef __attribute__((ext_vector_type(4)))  unsigned  u32x4;
typedef __attribute__((ext_vector_type(8)))  int       i32x8;
typedef __attribute__((ext_vector_type(4)))  int       i32x4;

union FragAB { v16bf v; uint4 u[2]; };

__device__ __forceinline__ unsigned short f2bf(float f) {
  unsigned u = __float_as_uint(f);
  u += 0x7FFFu + ((u >> 16) & 1u);          // round-to-nearest-even
  return (unsigned short)(u >> 16);
}
__device__ __forceinline__ float bf2f(unsigned short h) {
  return __uint_as_float(((unsigned)h) << 16);
}

// ---------------------------------------------------------------------------
// init: zero counts + loss accumulator
// ---------------------------------------------------------------------------
__global__ void k_init(unsigned* __restrict__ cnt, float* __restrict__ lossAcc) {
  int i = blockIdx.x * blockDim.x + threadIdx.x;
  if (i < NCODE) cnt[i] = 0u;
  if (i == 0)    *lossAcc = 0.0f;
}

// ---------------------------------------------------------------------------
// prep_x: NCHW f32 -> token-major [NTOK][DIM] bf16 hi/lo (LDS tile transpose)
// grid = (32 hw-tiles, 8 c-tiles, 16 batch), block = 256
// ---------------------------------------------------------------------------
__global__ void k_prep_x(const float* __restrict__ in,
                         unsigned short* __restrict__ xh,
                         unsigned short* __restrict__ xl) {
  __shared__ float tile[32][33];
  const int hw0 = blockIdx.x * 32, c0 = blockIdx.y * 32, b = blockIdx.z;
  const int tid = threadIdx.x;
#pragma unroll
  for (int i = 0; i < 4; ++i) {
    int e = tid + 256 * i, cl = e >> 5, wl = e & 31;
    tile[cl][wl] = in[(size_t)b * DIM * HW + (size_t)(c0 + cl) * HW + hw0 + wl];
  }
  __syncthreads();
#pragma unroll
  for (int i = 0; i < 4; ++i) {
    int e = tid + 256 * i, tl = e >> 5, cl = e & 31;
    float v = tile[cl][tl];
    unsigned short hi = f2bf(v);
    unsigned short lo = f2bf(v - bf2f(hi));
    size_t o = (size_t)(b * HW + hw0 + tl) * DIM + c0 + cl;
    xh[o] = hi;  xl[o] = lo;
  }
}

// ---------------------------------------------------------------------------
// prep_w: weight rows -> bf16 hi/lo + squared row norms. grid = NCODE, blk 256
// ---------------------------------------------------------------------------
__global__ void k_prep_w(const float* __restrict__ w,
                         unsigned short* __restrict__ wh,
                         unsigned short* __restrict__ wl,
                         float* __restrict__ enorm) {
  __shared__ float red[8];
  const int k = blockIdx.x, tid = threadIdx.x;
  float v = w[(size_t)k * DIM + tid];
  unsigned short hi = f2bf(v);
  wh[(size_t)k * DIM + tid] = hi;
  wl[(size_t)k * DIM + tid] = f2bf(v - bf2f(hi));
  float s = v * v;
#pragma unroll
  for (int off = 16; off > 0; off >>= 1) s += __shfl_xor(s, off, 32);
  if ((tid & 31) == 0) red[tid >> 5] = s;
  __syncthreads();
  if (tid == 0) {
    float t = 0.f;
#pragma unroll
    for (int i = 0; i < 8; ++i) t += red[i];
    enorm[k] = t;
  }
}

// ---------------------------------------------------------------------------
// TDM: issue one 32-row x 256-col bf16 tile (16 KB) global -> LDS.
// D# per CDNA5 ISA 8.3/8.4: data_size=1(2B), tensor_dim0=256, stride=256,
// tile = 256 x 32, type=2 in group0 bits 127:126.
// clang-23 toolchain: 6-arg builtin (g0,u32x4; g1,i32x8; g2,i32x4; g3,i32x4;
// extra i32x8; cpol).
// ---------------------------------------------------------------------------
__device__ __forceinline__ void tdm_issue(unsigned lds_addr,
                                          const unsigned short* gptr) {
#if USE_TDM
  unsigned long long ga = (unsigned long long)(size_t)gptr;
  u32x4 g0 = { 1u,                                   // count=1
               lds_addr,                             // lds byte address
               (unsigned)(ga & 0xFFFFFFFFu),         // global_addr lo
               (unsigned)((ga >> 32) & 0x01FFFFFFu) | 0x80000000u }; // hi|type=2
  i32x8 g1 = { (int)0x00010000,      // workgroup_mask=0, data_size=1 (2 bytes)
               (int)(256u << 16),    // tensor_dim0 = 256 (bits 79:48 lo half)
               (int)(32u  << 16),    // tensor_dim1 = 32  (bits 111:80 lo half)
               (int)(256u << 16),    // tile_dim0   = 256 (bits 127:112)
               (int)32,              // tile_dim1 = 32, tile_dim2 = 0
               (int)256,             // tensor_dim0_stride = 256
               0, 0 };
  i32x4 gz4  = { 0, 0, 0, 0 };
  i32x8 gz8  = { 0, 0, 0, 0, 0, 0, 0, 0 };
  __builtin_amdgcn_tensor_load_to_lds(g0, g1, gz4, gz4, gz8, 0);
#else
  (void)lds_addr; (void)gptr;
#endif
}

// ---------------------------------------------------------------------------
// argmin core. grid = NTOK/64, block = 256 (8 waves = 4 M-subtiles x 2 N).
// Each wave: A = 16 tokens x 256 dims (hi+lo) pinned in VGPRs; loops all 8192
// codes in tiles of 32 staged in LDS (double buffered via TDM).
// Per K-step: 3x v_wmma_f32_16x16x32_bf16 (hi*hi + hi*lo + lo*hi).
// ---------------------------------------------------------------------------
__global__ void __launch_bounds__(256)
k_argmin(const unsigned short* __restrict__ xh,
         const unsigned short* __restrict__ xl,
         const unsigned short* __restrict__ wh,
         const unsigned short* __restrict__ wl,
         const float* __restrict__ enorm,
         int* __restrict__ bestIdx) {
  __shared__ unsigned short shB[2][2][32 * DIM];      // 64 KB: [buf][hi/lo]
  const int tid  = threadIdx.x;
  const int lane = tid & 31, wid = tid >> 5;
  const int wm = wid >> 1, wn = wid & 1;
  const int lr = lane & 15, lh = lane >> 4;

  // -- load A fragments (this wave's 16 tokens), keep resident in VGPRs -----
  FragAB ahi[8], alo[8];
  {
    const size_t rowbase = (size_t)(blockIdx.x * 64 + wm * 16 + lr) * DIM;
    const int offa = lh * 8;        // lanes 0-15: K 0..7/16..23; 16-31: +8
#pragma unroll
    for (int s = 0; s < 8; ++s) {
      const int db = s * 32;
      ahi[s].u[0] = *(const uint4*)(xh + rowbase + db + offa);
      ahi[s].u[1] = *(const uint4*)(xh + rowbase + db + offa + 16);
      alo[s].u[0] = *(const uint4*)(xl + rowbase + db + offa);
      alo[s].u[1] = *(const uint4*)(xl + rowbase + db + offa + 16);
    }
  }

  float bv[8]; int bi[8];
#pragma unroll
  for (int r = 0; r < 8; ++r) { bv[r] = 3.4e38f; bi[r] = 0; }

  const unsigned ldsBase = (unsigned)(size_t)(&shB[0][0][0]);
  const int NIT = NCODE / 32;       // 256 code tiles

  if (wid == 0) {                   // prologue: stage tile 0 into buffer 0
    tdm_issue(ldsBase,          wh);
    tdm_issue(ldsBase + 16384u, wl);
  }

  for (int it = 0; it < NIT; ++it) {
    const int buf = it & 1;
    __syncthreads();                // previous compute done -> other buf free
    if (wid == 0) {
      if (it + 1 < NIT) {           // prefetch next tile, then wait for cur
        const unsigned dst = ldsBase + (unsigned)((it + 1) & 1) * 32768u;
        const size_t go = (size_t)(it + 1) * 32 * DIM;
        tdm_issue(dst,          wh + go);
        tdm_issue(dst + 16384u, wl + go);
        __builtin_amdgcn_s_wait_tensorcnt(2);
      } else {
        __builtin_amdgcn_s_wait_tensorcnt(0);
      }
    }
    __syncthreads();                // tile `it` visible to all waves

    const int c0 = it * 32;
    const int codeLane = c0 + wn * 16 + lr;   // this lane's column (code id)
    const float en = enorm[codeLane];

    v8f acc = { 0.f, 0.f, 0.f, 0.f, 0.f, 0.f, 0.f, 0.f };
    const unsigned short* bh = &shB[buf][0][0];
    const unsigned short* bl = &shB[buf][1][0];
    const int boff = (wn * 16 + lr) * DIM + lh * 8;   // B: lanes<16 K0..7/16..23; 16-31: +8
#pragma unroll
    for (int s = 0; s < 8; ++s) {
      FragAB fbh, fbl;
      const uint4* ph = (const uint4*)(bh + boff + s * 32);
      fbh.u[0] = ph[0]; fbh.u[1] = ph[2];
      const uint4* pl = (const uint4*)(bl + boff + s * 32);
      fbl.u[0] = pl[0]; fbl.u[1] = pl[2];
      acc = __builtin_amdgcn_wmma_f32_16x16x32_bf16(false, ahi[s].v, false, fbh.v,
                                                    (short)0, acc, false, false);
      acc = __builtin_amdgcn_wmma_f32_16x16x32_bf16(false, ahi[s].v, false, fbl.v,
                                                    (short)0, acc, false, false);
      acc = __builtin_amdgcn_wmma_f32_16x16x32_bf16(false, alo[s].v, false, fbh.v,
                                                    (short)0, acc, false, false);
    }
    // distance proxy d = ||e||^2 - 2 x.e ; per-lane running argmin
#pragma unroll
    for (int r = 0; r < 8; ++r) {
      float d = fmaf(-2.f, acc[r], en);
      if (d < bv[r]) { bv[r] = d; bi[r] = codeLane; }
    }
  }

  // cross-lane argmin over the 16 lanes sharing each row (xor tree, half-wave)
#pragma unroll
  for (int r = 0; r < 8; ++r) {
    float v = bv[r]; int ix = bi[r];
#pragma unroll
    for (int off = 8; off > 0; off >>= 1) {
      float ov = __shfl_xor(v, off, 32);
      int   oi = __shfl_xor(ix, off, 32);
      if (ov < v || (ov == v && oi < ix)) { v = ov; ix = oi; }
    }
    bv[r] = v; bi[r] = ix;
  }
  if (lr == 0) {                    // lanes 0 / 16 write rows 0-7 / 8-15
    const int t0 = blockIdx.x * 64 + wm * 16 + lh * 8;
#pragma unroll
    for (int r = 0; r < 8; ++r) bestIdx[t0 + r] = bi[r];
  }
}

// ---------------------------------------------------------------------------
// quantize + loss + counts. grid = NTOK/32, block = 256 (32 tok x 8 c-groups)
// ---------------------------------------------------------------------------
__global__ void k_quant(const float* __restrict__ in, const float* __restrict__ w,
                        const int* __restrict__ bestIdx, float* __restrict__ dout,
                        unsigned* __restrict__ cnt, float* __restrict__ lossAcc) {
  __shared__ float red[8];
  const int tid = threadIdx.x;
  const int t  = blockIdx.x * 32 + (tid & 31);
  const int cg = tid >> 5;
  const int idx = bestIdx[t];
  const int b = t >> 10, hw = t & 1023;
  const size_t ibase = (size_t)b * DIM * HW + hw;
  float accL = 0.f;
#pragma unroll 4
  for (int j = 0; j < 32; ++j) {
    const int c = cg * 32 + j;
    float wv = w[(size_t)idx * DIM + c];
    float xv = in[ibase + (size_t)c * HW];
    dout[1 + ibase + (size_t)c * HW] = wv;        // straight-through fwd = q
    float d = wv - xv;
    accL = fmaf(d, d, accL);
  }
  if (tid < 32) atomicAdd(&cnt[idx], 1u);
#pragma unroll
  for (int off = 16; off > 0; off >>= 1) accL += __shfl_xor(accL, off, 32);
  if ((tid & 31) == 0) red[tid >> 5] = accL;
  __syncthreads();
  if (tid == 0) {
    float s = 0.f;
#pragma unroll
    for (int i = 0; i < 8; ++i) s += red[i];
    atomicAdd(lossAcc, s);
  }
}

// ---------------------------------------------------------------------------
// finalize: loss scalar, perplexity, indices-as-float. 1 block x 256
// ---------------------------------------------------------------------------
__global__ void k_final(const unsigned* __restrict__ cnt,
                        const float* __restrict__ lossAcc,
                        const int* __restrict__ bestIdx,
                        float* __restrict__ dout) {
  __shared__ float red[8];
  const int tid = threadIdx.x;
  float s = 0.f;
  for (int k = tid; k < NCODE; k += 256) {
    float p = (float)cnt[k] * (1.0f / (float)NTOK);
    s += p * logf(p + 1e-10f);
  }
#pragma unroll
  for (int off = 16; off > 0; off >>= 1) s += __shfl_xor(s, off, 32);
  if ((tid & 31) == 0) red[tid >> 5] = s;
  __syncthreads();
  if (tid == 0) {
    float t = 0.f;
#pragma unroll
    for (int i = 0; i < 8; ++i) t += red[i];
    dout[1 + NELEM] = expf(-t);                          // perplexity
    dout[0] = lossAcc[0] * 1.25f / (float)NELEM;         // q + 0.25*e loss
  }
  for (int t2 = tid; t2 < NTOK; t2 += 256)
    dout[2 + NELEM + t2] = (float)bestIdx[t2];           // indices
}

// ---------------------------------------------------------------------------
extern "C" void kernel_launch(void* const* d_in, const int* in_sizes, int n_in,
                              void* d_out, int out_size, void* d_ws, size_t ws_size,
                              hipStream_t stream) {
  (void)in_sizes; (void)n_in; (void)out_size; (void)ws_size;
  const float* in = (const float*)d_in[0];        // [16,256,32,32] f32
  const float* w  = (const float*)d_in[1];        // [8192,256]    f32
  float* dout = (float*)d_out;

  char* ws = (char*)d_ws;
  unsigned short* xh = (unsigned short*)(ws);                       // 8 MB
  unsigned short* xl = (unsigned short*)(ws + 8388608);             // 8 MB
  unsigned short* wh = (unsigned short*)(ws + 16777216);            // 4 MB
  unsigned short* wl = (unsigned short*)(ws + 20971520);            // 4 MB
  float*    enorm    = (float*)(ws + 25165824);                     // 32 KB
  int*      bestIdx  = (int*)(ws + 25165824 + 32768);               // 64 KB
  unsigned* cnt      = (unsigned*)(ws + 25165824 + 32768 + 65536);  // 32 KB
  float*    lossAcc  = (float*)(ws + 25165824 + 32768 + 65536 + 32768);

  k_init<<<32, 256, 0, stream>>>(cnt, lossAcc);
  dim3 gpx(32, 8, 16);
  k_prep_x<<<gpx, 256, 0, stream>>>(in, xh, xl);
  k_prep_w<<<NCODE, 256, 0, stream>>>(w, wh, wl, enorm);
  k_argmin<<<NTOK / 64, 256, 0, stream>>>(xh, xl, wh, wl, enorm, bestIdx);
  k_quant<<<NTOK / 32, 256, 0, stream>>>(in, w, bestIdx, dout, cnt, lossAcc);
  k_final<<<1, 256, 0, stream>>>(cnt, lossAcc, bestIdx, dout);
}
